// NetworkInNetwork_23184233463819
// MI455X (gfx1250) — compile-verified, matrix operand order checked by reading, not run on hardware
//
#include <hip/hip_runtime.h>
#include <math.h>

typedef _Float16 v16h __attribute__((ext_vector_type(16)));
typedef _Float16 h2   __attribute__((ext_vector_type(2)));
typedef float    v8f  __attribute__((ext_vector_type(8)));

#define D_IN   64
#define D_OUT  64
#define GN_EPS 1e-5f

// Map a (k, col) element of a K x 64 weight matrix into the WMMA B-fragment
// swizzled LDS layout: [ntile][kstep][lane][j], 16 halves contiguous per lane.
// 16-bit fragment layout (ISA 7.12.2): lanes 0-15 hold K offsets {0..7,16..23},
// lanes 16-31 hold {8..15,24..31} within each 32-wide k-step.
__device__ __forceinline__ int bfrag_index(int k, int col, int nsteps) {
    const int s    = k >> 5;
    const int kr   = k & 31;
    const int hf   = (kr >> 3) & 1;                  // lane half
    const int j    = (kr & 7) + ((kr & 16) ? 8 : 0); // position in fragment
    const int lane = hf * 16 + (col & 15);
    const int n    = col >> 4;                       // ntile
    return (((n * nsteps + s) * 32) + lane) * 16 + j;
}

// Pack four float4 chunks into a 16-wide f16 A fragment register.
__device__ __forceinline__ void pack16(v16h& a, const float4 f0, const float4 f1,
                                       const float4 f2, const float4 f3) {
    a[0]  = (_Float16)f0.x; a[1]  = (_Float16)f0.y; a[2]  = (_Float16)f0.z; a[3]  = (_Float16)f0.w;
    a[4]  = (_Float16)f1.x; a[5]  = (_Float16)f1.y; a[6]  = (_Float16)f1.z; a[7]  = (_Float16)f1.w;
    a[8]  = (_Float16)f2.x; a[9]  = (_Float16)f2.y; a[10] = (_Float16)f2.z; a[11] = (_Float16)f2.w;
    a[12] = (_Float16)f3.x; a[13] = (_Float16)f3.y; a[14] = (_Float16)f3.z; a[15] = (_Float16)f3.w;
}

// ---------------------------------------------------------------------------
// Kernel 1: x2 = elu(x1 @ w1 + b1)   [nn,64] = [nn,64]@[64,64], x2 stored f16
// (halves the dominant edge-gather traffic; consistent with f16 WMMA inputs).
// 8 waves/block: 2 M-tiles x 4 N-tiles; w1 staged in LDS fragment-swizzled.
// ---------------------------------------------------------------------------
__global__ void gemm1_elu_kernel(const float* __restrict__ x1,
                                 const float* __restrict__ w1,
                                 const float* __restrict__ b1,
                                 _Float16* __restrict__ x2, int nn) {
    __shared__ _Float16 bfrag[4 * 2 * 32 * 16];        // 8 KB, fragment-swizzled
    for (int i = threadIdx.x; i < 64 * 64 / 4; i += 256) {
        const float4 v = ((const float4*)w1)[i];
        const int k  = (4 * i) >> 6;                   // row of w1 (K index)
        const int c0 = (4 * i) & 63;                   // col base
        bfrag[bfrag_index(k, c0 + 0, 2)] = (_Float16)v.x;
        bfrag[bfrag_index(k, c0 + 1, 2)] = (_Float16)v.y;
        bfrag[bfrag_index(k, c0 + 2, 2)] = (_Float16)v.z;
        bfrag[bfrag_index(k, c0 + 3, 2)] = (_Float16)v.w;
    }
    __syncthreads();

    const int lane  = threadIdx.x & 31;
    const int wv    = threadIdx.x >> 5;
    const int mtile = blockIdx.x * 2 + (wv >> 2);
    const int ntile = wv & 3;
    const int half  = lane >> 4;
    const int l16   = lane & 15;
    const int row   = mtile * 16 + l16;
    const int col   = ntile * 16 + l16;
    const int kbase = half * 8;
    // Clamp: out-of-range A rows only feed D rows we never store.
    const int rowc  = row < nn ? row : nn - 1;
    const float* arow = x1 + (size_t)rowc * D_IN;
    const v16h* bf = (const v16h*)bfrag;

    v8f c = {};
#pragma unroll
    for (int kk = 0; kk < 64; kk += 32) {
        v16h a;
        const float4 f0 = *(const float4*)(arow + kk + kbase);
        const float4 f1 = *(const float4*)(arow + kk + kbase + 4);
        const float4 f2 = *(const float4*)(arow + kk + kbase + 16);
        const float4 f3 = *(const float4*)(arow + kk + kbase + 20);
        pack16(a, f0, f1, f2, f3);
        const v16h b = bf[(ntile * 2 + (kk >> 5)) * 32 + lane];
        c = __builtin_amdgcn_wmma_f32_16x16x32_f16(false, a, false, b,
                                                   (short)0, c, false, false);
    }

    const float bias = b1[col];
    _Float16* orow0 = x2 + (size_t)(mtile * 16 + half * 8) * D_OUT + col;
    if ((mtile + 1) * 16 <= nn) {                      // full tile: unguarded
#pragma unroll
        for (int r = 0; r < 8; ++r) {
            float v = c[r] + bias;
            v = v > 0.f ? v : (__expf(v) - 1.f);       // ELU (alpha=1)
            orow0[(size_t)r * D_OUT] = (_Float16)v;
        }
    } else {                                           // tail tile
#pragma unroll
        for (int r = 0; r < 8; ++r) {
            const int orow = mtile * 16 + half * 8 + r;
            if (orow < nn) {
                float v = c[r] + bias;
                v = v > 0.f ? v : (__expf(v) - 1.f);
                x2[(size_t)orow * D_OUT + col] = (_Float16)v;
            }
        }
    }
}

// ---------------------------------------------------------------------------
// Kernel 2: agg[row] += x2[col] per edge. One wave per edge; each lane loads
// a packed f16 pair (b32) and issues two global_atomic_add_f32.
// ---------------------------------------------------------------------------
__global__ void scatter_add_kernel(const int* __restrict__ erow,
                                   const int* __restrict__ ecol,
                                   const _Float16* __restrict__ x2,
                                   float* __restrict__ agg, int ne) {
    const int e = blockIdx.x * 8 + (threadIdx.x >> 5);
    if (e >= ne) return;
    const int lane = threadIdx.x & 31;
    const int r = erow[e];
    const int c = ecol[e];
    const h2 v = *(const h2*)(x2 + (size_t)c * D_OUT + lane * 2);
    float* dst = agg + (size_t)r * D_OUT + lane * 2;
    __hip_atomic_fetch_add(dst,     (float)v[0], __ATOMIC_RELAXED, __HIP_MEMORY_SCOPE_AGENT);
    __hip_atomic_fetch_add(dst + 1, (float)v[1], __ATOMIC_RELAXED, __HIP_MEMORY_SCOPE_AGENT);
}

// ---------------------------------------------------------------------------
// Kernel 3a: per-feature sum / sum-of-squares over all nodes (float4 rows).
// ---------------------------------------------------------------------------
__global__ void col_stats_kernel(const float* __restrict__ agg,
                                 float* __restrict__ sums, int nn) {
    const int f4     = (threadIdx.x & 15) * 4;
    const int r0     = (int)(blockIdx.x * 16 + (threadIdx.x >> 4));
    const int stride = (int)(gridDim.x * 16);
    float4 s  = {0.f, 0.f, 0.f, 0.f};
    float4 s2 = {0.f, 0.f, 0.f, 0.f};
    for (int r = r0; r < nn; r += stride) {
        const float4 v = *(const float4*)(agg + (size_t)r * D_OUT + f4);
        s.x += v.x; s.y += v.y; s.z += v.z; s.w += v.w;
        s2.x += v.x * v.x; s2.y += v.y * v.y; s2.z += v.z * v.z; s2.w += v.w * v.w;
    }
    __hip_atomic_fetch_add(&sums[f4 + 0], s.x, __ATOMIC_RELAXED, __HIP_MEMORY_SCOPE_AGENT);
    __hip_atomic_fetch_add(&sums[f4 + 1], s.y, __ATOMIC_RELAXED, __HIP_MEMORY_SCOPE_AGENT);
    __hip_atomic_fetch_add(&sums[f4 + 2], s.z, __ATOMIC_RELAXED, __HIP_MEMORY_SCOPE_AGENT);
    __hip_atomic_fetch_add(&sums[f4 + 3], s.w, __ATOMIC_RELAXED, __HIP_MEMORY_SCOPE_AGENT);
    __hip_atomic_fetch_add(&sums[64 + f4 + 0], s2.x, __ATOMIC_RELAXED, __HIP_MEMORY_SCOPE_AGENT);
    __hip_atomic_fetch_add(&sums[64 + f4 + 1], s2.y, __ATOMIC_RELAXED, __HIP_MEMORY_SCOPE_AGENT);
    __hip_atomic_fetch_add(&sums[64 + f4 + 2], s2.z, __ATOMIC_RELAXED, __HIP_MEMORY_SCOPE_AGENT);
    __hip_atomic_fetch_add(&sums[64 + f4 + 3], s2.w, __ATOMIC_RELAXED, __HIP_MEMORY_SCOPE_AGENT);
}

// ---------------------------------------------------------------------------
// Kernel 3b: fold GraphNorm into per-feature affine y = x*scale + bias.
// var = E[x^2] - 2*(m*ms)*E[x] + (m*ms)^2
// ---------------------------------------------------------------------------
__global__ void finalize_norm_kernel(const float* __restrict__ sums,
                                     const float* __restrict__ gw,
                                     const float* __restrict__ gb,
                                     const float* __restrict__ gms,
                                     float* __restrict__ sb, int nn) {
    const int f = threadIdx.x;
    const float inv_n = 1.f / (float)nn;
    const float m   = sums[f] * inv_n;
    const float ex2 = sums[64 + f] * inv_n;
    const float ms  = m * gms[f];
    const float var = ex2 - 2.f * ms * m + ms * ms;
    const float sc  = gw[f] * rsqrtf(var + GN_EPS);
    sb[f]      = sc;
    sb[64 + f] = gb[f] - ms * sc;
}

// ---------------------------------------------------------------------------
// Kernel 4: out = [graphnorm(agg) | x1] @ w2 + b2. K=128: kk∈{0,32} read the
// normalized aggregate (affine fused into A load), kk∈{64,96} read x1.
// w2 staged in LDS pre-swizzled into B-fragment layout.
// ---------------------------------------------------------------------------
__global__ void gemm2_kernel(const float* __restrict__ agg,
                             const float* __restrict__ x1,
                             const float* __restrict__ sb,
                             const float* __restrict__ w2,
                             const float* __restrict__ b2,
                             float* __restrict__ out, int nn) {
    __shared__ _Float16 bfrag[4 * 4 * 32 * 16];        // 16 KB, fragment-swizzled
    for (int i = threadIdx.x; i < 128 * 64 / 4; i += 256) {
        const float4 v = ((const float4*)w2)[i];
        const int k  = (4 * i) >> 6;
        const int c0 = (4 * i) & 63;
        bfrag[bfrag_index(k, c0 + 0, 4)] = (_Float16)v.x;
        bfrag[bfrag_index(k, c0 + 1, 4)] = (_Float16)v.y;
        bfrag[bfrag_index(k, c0 + 2, 4)] = (_Float16)v.z;
        bfrag[bfrag_index(k, c0 + 3, 4)] = (_Float16)v.w;
    }
    __syncthreads();

    const int lane  = threadIdx.x & 31;
    const int wv    = threadIdx.x >> 5;
    const int mtile = blockIdx.x * 2 + (wv >> 2);
    const int ntile = wv & 3;
    const int half  = lane >> 4;
    const int l16   = lane & 15;
    const int row   = mtile * 16 + l16;
    const int col   = ntile * 16 + l16;
    const int kbase = half * 8;
    const int rowc  = row < nn ? row : nn - 1;
    const float* grow = agg + (size_t)rowc * 64;
    const float* xrow = x1 + (size_t)rowc * D_IN;
    const v16h* bf = (const v16h*)bfrag;

    v8f c = {};
#pragma unroll
    for (int kk = 0; kk < 128; kk += 32) {
        v16h a;
        float4 f0, f1, f2, f3;
        if (kk < 64) {
            // normalized aggregate: v*scale + bias, vectorized
            const int kb = kk + kbase;
            const float4 g0 = *(const float4*)(grow + kb);
            const float4 g1 = *(const float4*)(grow + kb + 4);
            const float4 g2 = *(const float4*)(grow + kb + 16);
            const float4 g3 = *(const float4*)(grow + kb + 20);
            const float4 s0 = *(const float4*)(sb + kb);
            const float4 s1 = *(const float4*)(sb + kb + 4);
            const float4 s2 = *(const float4*)(sb + kb + 16);
            const float4 s3 = *(const float4*)(sb + kb + 20);
            const float4 t0 = *(const float4*)(sb + 64 + kb);
            const float4 t1 = *(const float4*)(sb + 64 + kb + 4);
            const float4 t2 = *(const float4*)(sb + 64 + kb + 16);
            const float4 t3 = *(const float4*)(sb + 64 + kb + 20);
            f0 = make_float4(g0.x * s0.x + t0.x, g0.y * s0.y + t0.y,
                             g0.z * s0.z + t0.z, g0.w * s0.w + t0.w);
            f1 = make_float4(g1.x * s1.x + t1.x, g1.y * s1.y + t1.y,
                             g1.z * s1.z + t1.z, g1.w * s1.w + t1.w);
            f2 = make_float4(g2.x * s2.x + t2.x, g2.y * s2.y + t2.y,
                             g2.z * s2.z + t2.z, g2.w * s2.w + t2.w);
            f3 = make_float4(g3.x * s3.x + t3.x, g3.y * s3.y + t3.y,
                             g3.z * s3.z + t3.z, g3.w * s3.w + t3.w);
        } else {
            const int kb = (kk - 64) + kbase;
            f0 = *(const float4*)(xrow + kb);
            f1 = *(const float4*)(xrow + kb + 4);
            f2 = *(const float4*)(xrow + kb + 16);
            f3 = *(const float4*)(xrow + kb + 20);
        }
        pack16(a, f0, f1, f2, f3);
        const v16h b = bf[(ntile * 4 + (kk >> 5)) * 32 + lane];
        c = __builtin_amdgcn_wmma_f32_16x16x32_f16(false, a, false, b,
                                                   (short)0, c, false, false);
    }

    const float bias = b2[col];
    float* orow0 = out + (size_t)(mtile * 16 + half * 8) * D_OUT + col;
    if ((mtile + 1) * 16 <= nn) {                      // full tile: unguarded
#pragma unroll
        for (int r = 0; r < 8; ++r)
            orow0[(size_t)r * D_OUT] = c[r] + bias;
    } else {                                           // tail tile
#pragma unroll
        for (int r = 0; r < 8; ++r) {
            const int orow = mtile * 16 + half * 8 + r;
            if (orow < nn)
                out[(size_t)orow * D_OUT + col] = c[r] + bias;
        }
    }
}

// ---------------------------------------------------------------------------
extern "C" void kernel_launch(void* const* d_in, const int* in_sizes, int n_in,
                              void* d_out, int out_size, void* d_ws, size_t ws_size,
                              hipStream_t stream) {
    const float* x1  = (const float*)d_in[0];
    const int*   ei  = (const int*)d_in[1];   // [2, ne]
    const float* w1  = (const float*)d_in[2];
    const float* b1  = (const float*)d_in[3];
    const float* w2  = (const float*)d_in[4];
    const float* b2  = (const float*)d_in[5];
    const float* gw  = (const float*)d_in[6];
    const float* gb  = (const float*)d_in[7];
    const float* gms = (const float*)d_in[8];

    const int nn = in_sizes[0] / D_IN;
    const int ne = in_sizes[1] / 2;

    // Workspace layout (floats): [agg nn*64][sums 128][sb 128][x2 f16 nn*64]
    float* ws   = (float*)d_ws;
    float* agg  = ws;                                // nn*64 f32
    float* sums = agg + (size_t)nn * 64;             // 128 f32 (sum, sumsq)
    float* sb   = sums + 128;                        // 128 f32 (scale, bias)
    _Float16* x2 = (_Float16*)(sb + 128);            // nn*64 f16

    // Zero the aggregation buffer + stats each call (deterministic).
    hipMemsetAsync(agg, 0, ((size_t)nn * 64 + 128) * sizeof(float), stream);

    const int mtiles = (nn + 15) / 16;
    const dim3 blk(256);

    gemm1_elu_kernel<<<dim3((mtiles + 1) / 2), blk, 0, stream>>>(x1, w1, b1, x2, nn);
    scatter_add_kernel<<<dim3((ne + 7) / 8), blk, 0, stream>>>(ei, ei + ne, x2, agg, ne);
    col_stats_kernel<<<dim3(1024), blk, 0, stream>>>(agg, sums, nn);
    finalize_norm_kernel<<<dim3(1), dim3(64), 0, stream>>>(sums, gw, gb, gms, sb, nn);
    gemm2_kernel<<<dim3((mtiles + 1) / 2), blk, 0, stream>>>(agg, x1, sb, w2, b2,
                                                             (float*)d_out, nn);
}
